// SelfModel_76630806495730
// MI455X (gfx1250) — matrix operand order repaired and checked by reading, compiled, not run on hardware
//
#include <hip/hip_runtime.h>

// ---------------------------------------------------------------------------
// MI455X (gfx1250) implementation: GCN(2 layer) -> LSTM(28 steps) -> head.
// All matmuls via v_wmma_f32_16x16x32_bf16 (bf16 operands, f32 accum).
// Memory-bound on adjacency A (235MB x 3 passes ~= 35us at 23.3 TB/s);
// A is streamed with float4 loads, WMMA operands staged/packed in LDS.
// ---------------------------------------------------------------------------

typedef __attribute__((ext_vector_type(16))) __bf16          v16bf;
typedef __attribute__((ext_vector_type(8)))  float           v8f;
typedef __attribute__((ext_vector_type(16))) unsigned short  v16us;

constexpr int Bb = 8, Dd = 28, Nn = 512;
constexpr int ROWS = Bb * Dd * Nn;          // 114688 (b,d,n) rows

// ---- bf16 helpers ---------------------------------------------------------
__device__ inline unsigned short f2bf(float f) {
  unsigned u = __builtin_bit_cast(unsigned, f);
  u += 0x7FFFu + ((u >> 16) & 1u);          // round-to-nearest-even
  return (unsigned short)(u >> 16);
}
__device__ inline ushort4 f2bf4(float4 v) {
  return make_ushort4(f2bf(v.x), f2bf(v.y), f2bf(v.z), f2bf(v.w));
}
__device__ inline float bf2f(unsigned short h) {
  unsigned u = ((unsigned)h) << 16;
  return __builtin_bit_cast(float, u);
}
__device__ inline __bf16 us2bf(unsigned short h) {
  return __builtin_bit_cast(__bf16, h);
}
__device__ inline float sigm(float x) { return 1.0f / (1.0f + __expf(-x)); }

__device__ inline v8f wmma_bf16(v16bf a, v16bf b, v8f c) {
  // 8 args: (neg_a, A, neg_b, B, c_mod, C, reuse_a, reuse_b)
  return __builtin_amdgcn_wmma_f32_16x16x32_bf16(false, a, false, b,
                                                 (short)0, c, false, false);
}

// ---- WMMA fragment loaders (layouts per CDNA5 ISA 7.12.2) ------
// A (16x32 bf16): lane m = lane&15; element i -> K = ((i&8)<<1)|(hb<<3)|(i&7)
__device__ inline v16bf a_frag(const unsigned short* p, int ld, int lane) {
  const int m = lane & 15, hb = (lane >> 4) & 1;
  v16bf a;
#pragma unroll
  for (int i = 0; i < 16; ++i) {
    const int k = ((i & 8) << 1) | (hb << 3) | (i & 7);
    a[i] = us2bf(p[m * ld + k]);
  }
  return a;
}
__device__ inline v16bf a_frag_g(const unsigned short* p, int ld, int lane,
                                 int kbase, int kmax) {
  const int m = lane & 15, hb = (lane >> 4) & 1;
  v16bf a;
#pragma unroll
  for (int i = 0; i < 16; ++i) {
    const int k = kbase + (((i & 8) << 1) | (hb << 3) | (i & 7));
    a[i] = (k < kmax) ? us2bf(p[m * ld + k]) : us2bf((unsigned short)0);
  }
  return a;
}
// B operands are pre-packed in LDS in fragment order so every lane reads
// 16 contiguous bf16 (ds_load_b128 pair) per fragment.
// element (k, col): lane = ((k&8)?16:0) + (col&15), i = (k&7)|((k&16)>>1)
__device__ inline int bpack_pos(int k, int c, int KK) {
  return ((((c >> 4) * KK + (k >> 5)) * 32) + ((k & 8) ? 16 : 0) + (c & 15)) * 16 +
         ((k & 7) | ((k & 16) >> 1));
}
__device__ inline v16bf bfrag_packed(const unsigned short* p, int ct, int kk,
                                     int KK, int lane) {
  const v16us u = *(const v16us*)(p + (((ct * KK + kk) * 32) + lane) * 16);
  return __builtin_bit_cast(v16bf, u);
}

// ---------------------------------------------------------------------------
// K0: dinv[row] = rsqrt(1 + sum_m A[row,m])   (one wave per 512-elem row)
// ---------------------------------------------------------------------------
__global__ __launch_bounds__(256) void k_deg(const float* __restrict__ A,
                                             float* __restrict__ dinv) {
  const int wid = threadIdx.x >> 5, lane = threadIdx.x & 31;
  const int row = blockIdx.x * 8 + wid;
  const float4* ap4 = (const float4*)(A + (size_t)row * Nn);
  float s = 0.f;
#pragma unroll
  for (int m = lane; m < Nn / 4; m += 32) {
    const float4 v = ap4[m];
    s += (v.x + v.y) + (v.z + v.w);
  }
#pragma unroll
  for (int off = 16; off; off >>= 1) s += __shfl_xor(s, off, 32);
  if (lane == 0) {
    const float deg = s + 1.0f;
    dinv[row] = deg > 0.f ? rsqrtf(deg) : 0.f;
  }
}

// ---------------------------------------------------------------------------
// K1/K3: out[row,:] = bf16( dinv[row] * (src[row,:] @ W) )   rows=ROWS
// ---------------------------------------------------------------------------
template <int K, int NC, bool SRCBF>
__global__ __launch_bounds__(256) void k_rowgemm(const void* __restrict__ src,
                                                 const float* __restrict__ W,
                                                 const float* __restrict__ dinvp,
                                                 unsigned short* __restrict__ outBf) {
  constexpr int KK = K / 32, CT = NC / 16, LD = K + 4, KV = K / 4;
  __shared__ alignas(32) unsigned short sIn[128 * LD];
  __shared__ alignas(32) unsigned short sW[CT * KK * 32 * 16];
  const int tid = threadIdx.x, lane = tid & 31, w = tid >> 5;
  const int rbase = blockIdx.x * 128;

  for (int idx = tid; idx < K * NC; idx += 256) {
    const int k = idx / NC, c = idx % NC;
    sW[bpack_pos(k, c, KK)] = f2bf(W[idx]);
  }
  for (int idx = tid; idx < 128 * KV; idx += 256) {
    const int r = idx / KV, k4 = (idx % KV) * 4;
    if constexpr (SRCBF) {
      const ushort4 u = *(const ushort4*)((const unsigned short*)src +
                                          (size_t)(rbase + r) * K + k4);
      *(ushort4*)&sIn[r * LD + k4] = u;               // already bf16
    } else {
      const float4 v = *(const float4*)((const float*)src +
                                        (size_t)(rbase + r) * K + k4);
      *(ushort4*)&sIn[r * LD + k4] = f2bf4(v);
    }
  }
  __syncthreads();

  const unsigned short* myIn = sIn + (w * 16) * LD;
  v8f acc[CT];
#pragma unroll
  for (int ct = 0; ct < CT; ++ct)
#pragma unroll
    for (int e = 0; e < 8; ++e) acc[ct][e] = 0.f;

#pragma unroll
  for (int kk = 0; kk < KK; ++kk) {
    const v16bf a = a_frag(myIn + kk * 32, LD, lane);
#pragma unroll
    for (int ct = 0; ct < CT; ++ct)
      acc[ct] = wmma_bf16(a, bfrag_packed(sW, ct, kk, KK, lane), acc[ct]);
  }
  const int hi = lane >> 4, nn = lane & 15;
#pragma unroll
  for (int ct = 0; ct < CT; ++ct)
#pragma unroll
    for (int r = 0; r < 8; ++r) {
      const int row = rbase + w * 16 + hi * 8 + r;
      outBf[(size_t)row * NC + ct * 16 + nn] = f2bf(acc[ct][r] * dinvp[row]);
    }
}

// ---------------------------------------------------------------------------
// K2/K4: per (b,d):  v = dinv_n*(A@Y + Y) + bias
//   layer1: outH = relu(v) as bf16
//   layer2: seq[(b*N+n)*D + d][0..31] = v;  seq[...][32] = X[b,d,n,7]
// ---------------------------------------------------------------------------
template <int NC, bool LAYER2>
__global__ __launch_bounds__(256) void k_adjgemm(
    const float* __restrict__ A, const unsigned short* __restrict__ Y,
    const float* __restrict__ dinvp, const float* __restrict__ bias,
    unsigned short* __restrict__ outH, unsigned short* __restrict__ seq,
    const float* __restrict__ X) {
  constexpr int CT = NC / 16;
  __shared__ alignas(32) unsigned short sA[128 * 36];
  __shared__ alignas(32) unsigned short sY[CT * 32 * 16];
  const int tid = threadIdx.x, lane = tid & 31, w = tid >> 5;
  const int bd = blockIdx.x;              // 0..223
  const int rt = blockIdx.y * 128;        // n row-tile
  const float* Ap = A + (size_t)bd * Nn * Nn;
  const unsigned short* Yp = Y + (size_t)bd * Nn * NC;

  v8f acc[CT];
#pragma unroll
  for (int ct = 0; ct < CT; ++ct)
#pragma unroll
    for (int e = 0; e < 8; ++e) acc[ct][e] = 0.f;

  for (int kc = 0; kc < Nn / 32; ++kc) {
    const int kbase = kc * 32;
    for (int idx = tid; idx < 128 * 8; idx += 256) {     // A chunk, float4
      const int r = idx >> 3, k4 = (idx & 7) << 2;
      const float4 v = *(const float4*)&Ap[(size_t)(rt + r) * Nn + kbase + k4];
      *(ushort4*)&sA[r * 36 + k4] = f2bf4(v);
    }
    for (int idx = tid; idx < 32 * NC; idx += 256) {     // Y chunk, packed
      const int k = idx / NC, c = idx % NC;
      sY[bpack_pos(k, c, 1)] = Yp[(size_t)(kbase + k) * NC + c];
    }
    __syncthreads();
    const v16bf a = a_frag(sA + (w * 16) * 36, 36, lane);
#pragma unroll
    for (int ct = 0; ct < CT; ++ct)
      acc[ct] = wmma_bf16(a, bfrag_packed(sY, ct, 0, 1, lane), acc[ct]);
    __syncthreads();
  }

  const int hi = lane >> 4, nn = lane & 15;
  const int b = bd / Dd, d = bd % Dd;
#pragma unroll
  for (int ct = 0; ct < CT; ++ct) {
    const int c = ct * 16 + nn;
    const float bc = bias[c];
#pragma unroll
    for (int r = 0; r < 8; ++r) {
      const int n = rt + w * 16 + hi * 8 + r;
      const size_t gr = (size_t)bd * Nn + n;
      const float self = bf2f(Yp[(size_t)n * NC + c]);       // +I self loop
      const float v = dinvp[gr] * (acc[ct][r] + self) + bc;
      if constexpr (!LAYER2) {
        outH[gr * 64 + c] = f2bf(v > 0.f ? v : 0.f);
      } else {
        const size_t so = ((size_t)(b * Nn + n) * Dd + d) * 33;
        seq[so + c] = f2bf(v);
        if (ct == 0 && nn == 0) seq[so + 32] = f2bf(X[gr * 8 + 7]);  // X[...,-1]
      }
    }
  }
}

// ---------------------------------------------------------------------------
// K5: LSTM (28 steps, in=33, H=128) + head.
// Block = 256 threads = 2 seq-groups x 4 column-split waves; 32 seqs/block,
// 128 blocks.  Each wave owns 2 of the 8 hidden 16-col tiles and computes
// all four gates for them (48 wmma/step).  The 4 waves of a seq-group share
// the h buffer in LDS; per-step __syncthreads() orders producer/consumer.
// Weights packed fragment-order in LDS (~206KB < 320KB WGP LDS).
// Per-wave live state ~130 VGPRs -> no s_set_vgpr_msb churn, 2 waves/SIMD.
// ---------------------------------------------------------------------------
__global__ __launch_bounds__(256) void k_lstm(
    const unsigned short* __restrict__ seq, const float* __restrict__ Wi,
    const float* __restrict__ Wh, const float* __restrict__ bi,
    const float* __restrict__ bh, const float* __restrict__ Wfc,
    const float* __restrict__ bfc, float* __restrict__ out) {
  extern __shared__ __align__(64) unsigned short smem[];
  unsigned short* pWi  = smem;                    // 32ct*2kk*32*16 = 32768
  unsigned short* pWh  = pWi + 32 * 2 * 32 * 16;  // 32ct*4kk*32*16 = 65536
  unsigned short* pWfc = pWh + 32 * 4 * 32 * 16;  // 1ct*4kk*32*16  = 2048
  unsigned short* sX   = pWfc + 4 * 32 * 16;      // 2 groups * 16*34 = 1088
  unsigned short* sH   = sX + 2 * (16 * 34);      // 2 groups * 16*130 = 4160

  const int tid = threadIdx.x, lane = tid & 31, w = tid >> 5;
  const int grp = w >> 2, q = w & 3;   // seq-group, column-split index
  const int gtid = tid & 127;          // thread id within seq-group
  const int nn = lane & 15, hi = lane >> 4;

  for (int idx = tid; idx < 64 * 512; idx += 256) {  // Wi padded K 33->64
    const int k = idx >> 9, c = idx & 511;
    pWi[bpack_pos(k, c, 2)] = (k < 33) ? f2bf(Wi[k * 512 + c]) : (unsigned short)0;
  }
  for (int idx = tid; idx < 128 * 512; idx += 256) {
    const int k = idx >> 9, c = idx & 511;
    pWh[bpack_pos(k, c, 4)] = f2bf(Wh[k * 512 + c]);
  }
  for (int idx = tid; idx < 128 * 16; idx += 256) {  // Wfc padded N 14->16
    const int k = idx >> 4, c = idx & 15;
    pWfc[bpack_pos(k, c, 4)] = (c < 14) ? f2bf(Wfc[k * 14 + c]) : (unsigned short)0;
  }
  unsigned short* myX = sX + grp * (16 * 34);
  unsigned short* myH = sH + grp * (16 * 130);
  for (int i = gtid; i < 16 * 130; i += 128) myH[i] = 0;   // h0 = 0

  float bsum[4][2];                    // this wave's 2 hidden tiles x 4 gates
#pragma unroll
  for (int g = 0; g < 4; ++g)
#pragma unroll
    for (int ntl = 0; ntl < 2; ++ntl) {
      const int col = g * 128 + (q * 2 + ntl) * 16 + nn;
      bsum[g][ntl] = bi[col] + bh[col];
    }

  const int seqBase = blockIdx.x * 32 + grp * 16;
  v8f c8[2];
#pragma unroll
  for (int ntl = 0; ntl < 2; ++ntl)
#pragma unroll
    for (int e = 0; e < 8; ++e) c8[ntl][e] = 0.f;   // c0 = 0
  __syncthreads();

  for (int t = 0; t < Dd; ++t) {
    for (int idx = gtid; idx < 16 * 33; idx += 128) {
      const int s = idx / 33, f = idx % 33;
      myX[s * 34 + f] = seq[((size_t)(seqBase + s) * Dd + t) * 33 + f];
    }
    __syncthreads();
    const v16bf ax0 = a_frag(myX, 34, lane);              // x K=0..31
    const v16bf ax1 = a_frag_g(myX, 34, lane, 32, 33);    // x K=32 (padded)
    v16bf ah[4];
#pragma unroll
    for (int kk = 0; kk < 4; ++kk) ah[kk] = a_frag(myH + kk * 32, 130, lane);

    // this wave's 2 hidden tiles: compute i,f,g,o then update c and h
#pragma unroll
    for (int ntl = 0; ntl < 2; ++ntl) {
      const int nt = q * 2 + ntl;
      v8f ga[4];
#pragma unroll
      for (int g = 0; g < 4; ++g) {
        v8f a;
#pragma unroll
        for (int e = 0; e < 8; ++e) a[e] = bsum[g][ntl];
        const int ct = g * 8 + nt;                  // gate col-tile in 0..31
        a = wmma_bf16(ax0, bfrag_packed(pWi, ct, 0, 2, lane), a);
        a = wmma_bf16(ax1, bfrag_packed(pWi, ct, 1, 2, lane), a);
#pragma unroll
        for (int kk = 0; kk < 4; ++kk)
          a = wmma_bf16(ah[kk], bfrag_packed(pWh, ct, kk, 4, lane), a);
        ga[g] = a;
      }
#pragma unroll
      for (int e = 0; e < 8; ++e) {
        const float cc =
            sigm(ga[1][e]) * c8[ntl][e] + sigm(ga[0][e]) * tanhf(ga[2][e]);
        c8[ntl][e] = cc;
        myH[(hi * 8 + e) * 130 + nt * 16 + nn] =
            f2bf(sigm(ga[3][e]) * tanhf(cc));
      }
    }
    __syncthreads();
  }

  // head: out = h_last @ Wfc + bfc   (one wave per seq-group)
  if (q == 0) {
    v16bf ahf[4];
#pragma unroll
    for (int kk = 0; kk < 4; ++kk) ahf[kk] = a_frag(myH + kk * 32, 130, lane);
    v8f o_;
#pragma unroll
    for (int e = 0; e < 8; ++e) o_[e] = (nn < 14) ? bfc[nn] : 0.f;
#pragma unroll
    for (int kk = 0; kk < 4; ++kk)
      o_ = wmma_bf16(ahf[kk], bfrag_packed(pWfc, 0, kk, 4, lane), o_);
    if (nn < 14) {
#pragma unroll
      for (int r = 0; r < 8; ++r)
        out[(size_t)(seqBase + hi * 8 + r) * 14 + nn] = o_[r];
    }
  }
}

// ---------------------------------------------------------------------------
extern "C" void kernel_launch(void* const* d_in, const int* in_sizes, int n_in,
                              void* d_out, int out_size, void* d_ws,
                              size_t ws_size, hipStream_t stream) {
  const float* X   = (const float*)d_in[0];
  const float* A   = (const float*)d_in[1];
  const float* E   = (const float*)d_in[2];   // extra_info
  const float* W1  = (const float*)d_in[4];
  const float* b1  = (const float*)d_in[5];
  const float* W2  = (const float*)d_in[6];
  const float* b2  = (const float*)d_in[7];
  const float* Wi  = (const float*)d_in[8];
  const float* Wh  = (const float*)d_in[9];
  const float* bi  = (const float*)d_in[10];
  const float* bh  = (const float*)d_in[11];
  const float* Wfc = (const float*)d_in[12];
  const float* bfc = (const float*)d_in[13];
  float* out = (float*)d_out;

  // workspace carve-out (~45MB)
  char* ws = (char*)d_ws;
  size_t off = 0;
  float* dinv = (float*)(ws + off);            off += (size_t)ROWS * 4;
  unsigned short* Y1 = (unsigned short*)(ws + off); off += (size_t)ROWS * 64 * 2;
  unsigned short* Hb = (unsigned short*)(ws + off); off += (size_t)ROWS * 64 * 2;
  unsigned short* Y2 = (unsigned short*)(ws + off); off += (size_t)ROWS * 32 * 2;
  unsigned short* SQ = (unsigned short*)(ws + off); // [B*N][28][33] bf16

  k_deg<<<ROWS / 8, 256, 0, stream>>>(A, dinv);
  k_rowgemm<128, 64, false><<<ROWS / 128, 256, 0, stream>>>(E, W1, dinv, Y1);
  k_adjgemm<64, false><<<dim3(Bb * Dd, Nn / 128), 256, 0, stream>>>(
      A, Y1, dinv, b1, Hb, nullptr, nullptr);
  k_rowgemm<64, 32, true><<<ROWS / 128, 256, 0, stream>>>(Hb, W2, dinv, Y2);
  k_adjgemm<32, true><<<dim3(Bb * Dd, Nn / 128), 256, 0, stream>>>(
      A, Y2, dinv, b2, nullptr, SQ, X);

  const size_t lstm_smem =
      (size_t)(32 * 2 * 32 * 16 + 32 * 4 * 32 * 16 + 4 * 32 * 16 +
               2 * 16 * 34 + 2 * 16 * 130) * sizeof(unsigned short);
  k_lstm<<<Bb * Nn / 32, 256, lstm_smem, stream>>>(SQ, Wi, Wh, bi, bh, Wfc,
                                                   bfc, out);
}
